// CINLayer_39333310497371
// MI455X (gfx1250) — matrix operand order repaired and checked by reading, not compiled, not run on hardware
//
#include <hip/hip_runtime.h>
#include <stdint.h>

// CDNA5 / gfx1250 fused bilinear-contraction ("CIN layer") kernel.
//
//   output_mat[b,c,d] = sum_{i,j} kernel[c,i,j] * x[b,i,d] * y[b,j,d]
//   final_output[b,c] = sum_d output_mat[b,c,d]
//
// GEMM view: Out[m=(b,d), c] = Z[m, k=(i,j)] * Kflat[k, c], M=131072, N=64,
// K=1024. Z built on the fly (8x v_pk_mul_f16 per K=32 chunk); kernel
// pre-swizzled to f16 WMMA B-layout, staged ONCE per WG into 128 KB LDS and
// reused across 8 b-values (2 groups of 4). Outputs stored directly as b128
// (C/D layout gives each lane 8 consecutive d's).

typedef __attribute__((ext_vector_type(16))) _Float16 v16h;
typedef __attribute__((ext_vector_type(8)))  float    v8f;

#define B_TOT 4096
#define H     32
#define DD    32
#define CC    64
#define KF16_WORDS 32768   // 32 chunks * 4 n-tiles * 32 lanes * 8 packed words
#define BPG   4            // b per group (8 waves, 2 per b)
#define GROUPS 2           // groups per WG -> 8 b per WG, grid = 512

// ---------------------------------------------------------------------------
// Prep: kernel f32 [C,H,H] -> f16 packed words in WMMA B-matrix lane layout.
// Word w = ((i*4 + t)*32 + lane)*8 + v holds {k[c,i,jlo], k[c,i,jlo+1]},
// c = t*16 + (lane&15), jlo = ((lane>>4)<<4) + 2v.  (B 32x16 f16: lane%16 =
// column; this code's contiguous k order: lanes 0-15 K=0..15 packed, lanes 16-31 K=16..31, as in A.)
// ---------------------------------------------------------------------------
__global__ __launch_bounds__(256) void cin_prep_kernel(
    const float* __restrict__ kern, uint32_t* __restrict__ kf16) {
  int w = blockIdx.x * 256 + threadIdx.x;
  if (w >= KF16_WORDS) return;
  int v  = w & 7;
  int L  = (w >> 3) & 31;
  int ti = w >> 8;              // i*4 + t
  int t  = ti & 3;
  int i  = ti >> 2;
  int c  = t * 16 + (L & 15);
  int j  = ((L >> 4) << 4) + 2 * v;
  const float* src = kern + (c * H + i) * H + j;
  union { _Float16 h[2]; uint32_t u; } pk;
  pk.h[0] = (_Float16)src[0];
  pk.h[1] = (_Float16)src[1];
  kf16[w] = pk.u;
}

// ---------------------------------------------------------------------------
// Main kernel: 256 threads = 8 waves. Wave = 16 d-rows of one b, all 64 c.
// WG stages full 128 KB B once, then processes 2 groups of 4 b.
// ---------------------------------------------------------------------------
__global__ __launch_bounds__(256) void cin_main_kernel(
    const float* __restrict__ x, const float* __restrict__ y,
    const uint32_t* __restrict__ kf16, float* __restrict__ out) {
  __shared__ __align__(32) uint32_t ldsB[KF16_WORDS];  // 128 KB, resident B
  __shared__ float ldsR[2 * BPG][CC];                  // 2 KB reduction slots

  const int tid   = threadIdx.x;
  const int lane  = tid & 31;
  const int wave  = tid >> 5;
  const int laneM = lane & 15;
  const int hi16  = lane >> 4;
  const int b_loc = wave >> 1;          // 0..3 within group
  const int d0    = (wave & 1) << 4;    // 0 or 16
  const int j0    = hi16 ? 16 : 0;

  // ---- Stage the whole swizzled B operand into LDS (once per WG).
  {
    const uint4* src = (const uint4*)kf16;
    uint4* dst = (uint4*)ldsB;
#pragma unroll
    for (int k = 0; k < 32; ++k) dst[tid + k * 256] = src[tid + k * 256];
  }
  __syncthreads();

  const uint32_t* bp0 = ldsB + lane * 8;

  for (int g = 0; g < GROUPS; ++g) {
    const int b = blockIdx.x * (GROUPS * BPG) + g * BPG + b_loc;
    const int d = d0 + laneM;

    // ---- Gather x/y for this group: issue all loads first (clauseable),
    // convert afterwards, so HBM/L2 latency is pipelined, not serialized.
    const float* yb = y + (size_t)b * (H * DD) + d;
    const float* xb = x + (size_t)b * (H * DD) + d;
    float yf[16];
#pragma unroll
    for (int e = 0; e < 8; ++e) yf[e]     = yb[(j0 + e) * DD];
#pragma unroll
    for (int e = 0; e < 8; ++e) yf[8 + e] = yb[(j0 + 8 + e) * DD];
    float xf[32];
#pragma unroll
    for (int i = 0; i < 32; ++i) xf[i] = xb[i * DD];

    // One prefetch per operand pulls the next group's whole 4 KB slab
    // (lane*128B spacing covers 32 cachelines) -> global_prefetch_b8.
    if (g + 1 < GROUPS) {
      const int bn = blockIdx.x * (GROUPS * BPG) + (g + 1) * BPG + b_loc;
      __builtin_prefetch(x + (size_t)bn * (H * DD) + lane * 32, 0, 1);
      __builtin_prefetch(y + (size_t)bn * (H * DD) + lane * 32, 0, 1);
    }

    // y[b,:,d] as WMMA A operand (this code's contiguous k order, 16x32: lanes 0-15 hold
    // K=0..15, lanes 16-31 hold K=16..31, matching B; row m = lane%16 = d).
    v16h yv;
#pragma unroll
    for (int e = 0; e < 16; ++e) yv[e] = (_Float16)yf[e];
    _Float16 xh[32];
#pragma unroll
    for (int i = 0; i < 32; ++i) xh[i] = (_Float16)xf[i];

    v8f acc[4] = {v8f{}, v8f{}, v8f{}, v8f{}};

    // A triple-buffer: the pk_mul group writing abuf[(i+1)%3] is always >=4
    // WMMAs after the last WMMA that read that buffer.
    v16h abuf[3];
    abuf[0] = yv * xh[0];

#pragma unroll
    for (int i = 0; i < 32; ++i) {
      const uint32_t* bp = bp0 + i * 1024;
      v16h b0 = *(const v16h*)(bp);
      v16h b1 = *(const v16h*)(bp + 256);
      v16h b2 = *(const v16h*)(bp + 512);
      v16h b3 = *(const v16h*)(bp + 768);
      if (i + 1 < 32) abuf[(i + 1) % 3] = yv * xh[i + 1];
      acc[0] = __builtin_amdgcn_wmma_f32_16x16x32_f16(
          false, abuf[i % 3], false, b0, (short)0, acc[0], false, false);
      acc[1] = __builtin_amdgcn_wmma_f32_16x16x32_f16(
          false, abuf[i % 3], false, b1, (short)0, acc[1], false, false);
      acc[2] = __builtin_amdgcn_wmma_f32_16x16x32_f16(
          false, abuf[i % 3], false, b2, (short)0, acc[2], false, false);
      acc[3] = __builtin_amdgcn_wmma_f32_16x16x32_f16(
          false, abuf[i % 3], false, b3, (short)0, acc[3], false, false);
    }

    // ---- Direct coalesced stores: C/D layout => lane holds c = t*16+lane%16,
    // d = d0 + 8*hi16 + g for g=0..7 (8 consecutive floats) -> 2x b128.
    float* op = out + (size_t)b * (CC * DD) + d0 + (hi16 << 3);
#pragma unroll
    for (int t = 0; t < 4; ++t) {
      const int c = t * 16 + laneM;
      float4 lo = {acc[t][0], acc[t][1], acc[t][2], acc[t][3]};
      float4 hi = {acc[t][4], acc[t][5], acc[t][6], acc[t][7]};
      *(float4*)(op + c * DD)     = lo;
      *(float4*)(op + c * DD + 4) = hi;
    }

    // ---- final_output[b,c] = sum_d. Per-lane horizontal add (8 d's), fold
    // lane L with L+16 (other 8 d's of this wave), stash per-wave partials in
    // LDS, combine the two waves of each b.
#pragma unroll
    for (int t = 0; t < 4; ++t) {
      float s = acc[t][0] + acc[t][1] + acc[t][2] + acc[t][3] +
                acc[t][4] + acc[t][5] + acc[t][6] + acc[t][7];
      s += __shfl_xor(s, 16, 32);
      if (lane < 16) ldsR[wave][t * 16 + laneM] = s;
    }
    __syncthreads();
    if (tid < BPG * CC) {
      const int bl = tid >> 6;     // 0..3
      const int c  = tid & 63;
      float fin = ldsR[2 * bl][c] + ldsR[2 * bl + 1][c];
      out[(size_t)B_TOT * CC * DD +
          (size_t)(blockIdx.x * (GROUPS * BPG) + g * BPG + bl) * CC + c] = fin;
    }
    __syncthreads();   // ldsR reused next group
  }
}

// ---------------------------------------------------------------------------
extern "C" void kernel_launch(void* const* d_in, const int* in_sizes, int n_in,
                              void* d_out, int out_size, void* d_ws, size_t ws_size,
                              hipStream_t stream) {
  (void)in_sizes; (void)n_in; (void)out_size; (void)ws_size;
  const float* x    = (const float*)d_in[0];
  const float* y    = (const float*)d_in[1];
  const float* kern = (const float*)d_in[2];
  uint32_t* kf16 = (uint32_t*)d_ws;            // 128 KB scratch
  float* out = (float*)d_out;

  cin_prep_kernel<<<KF16_WORDS / 256, 256, 0, stream>>>(kern, kf16);
  cin_main_kernel<<<B_TOT / (GROUPS * BPG), 256, 0, stream>>>(x, y, kf16, out);
}
